// TransformerBlock_24180665876680
// MI455X (gfx1250) — matrix operand order repaired and checked by reading, compiled, not run on hardware
//
#include <hip/hip_runtime.h>

typedef __bf16 bf16_t;
typedef __bf16 v16bf __attribute__((ext_vector_type(16)));
typedef float  v8f   __attribute__((ext_vector_type(8)));

// --- Tensor Data Mover availability / arity detection (probe-documented) ---
#if defined(__gfx1250__) && __has_builtin(__builtin_amdgcn_tensor_load_to_lds)
#define USE_TDM 1
#if __has_include(<hip/amd_detail/amd_gfx1250_TDM.h>)
#define TDM_SIX_ARGS 1   // amdgpu-toolchain (clang-23 / therock-10.0 headers)
#else
#define TDM_SIX_ARGS 0   // ROCm 7.2 (clang-22): 5-arg builtin
#endif
#else
#define USE_TDM 0
#endif

struct __align__(16) Pack32 { uint4 lo; uint4 hi; };
union FragU { Pack32 p; v16bf v; };
union PackE { Pack32 p; bf16_t e[16]; };

__device__ __forceinline__ bf16_t f2bf(float f) {
  unsigned u = __builtin_bit_cast(unsigned, f);
  unsigned r = (u + 0x7fffu + ((u >> 16) & 1u)) >> 16;
  unsigned short h = (unsigned short)r;
  return __builtin_bit_cast(bf16_t, h);
}

__device__ __forceinline__ v16bf make_frag(uint4 lo, uint4 hi) {
  FragU u; u.p.lo = lo; u.p.hi = hi; return u.v;
}
__device__ __forceinline__ v16bf ld_frag_contig(const bf16_t* p) {
  return make_frag(*(const uint4*)p, *(const uint4*)(p + 8));
}
__device__ __forceinline__ v16bf ld_frag_split(const bf16_t* p0, const bf16_t* p1) {
  return make_frag(*(const uint4*)p0, *(const uint4*)p1);
}

__device__ __forceinline__ float redmax16(float x) {
#pragma unroll
  for (int m = 1; m < 16; m <<= 1) x = fmaxf(x, __shfl_xor(x, m, 32));
  return x;
}
__device__ __forceinline__ float redsum16(float x) {
#pragma unroll
  for (int m = 1; m < 16; m <<= 1) x += __shfl_xor(x, m, 32);
  return x;
}

#if USE_TDM
typedef unsigned int u32x4 __attribute__((ext_vector_type(4)));
typedef int i32x4 __attribute__((ext_vector_type(4)));
typedef int i32x8 __attribute__((ext_vector_type(8)));

// Issue a 2D tile DMA (global -> LDS) via the Tensor Data Mover.
// D# packed per CDNA5 ISA ch.8: group0 {count=1, lds_addr, global_addr, type=2},
// group1 {data_size=2B, tensor dims, tile dims, dim0 stride}. Groups 2/3 zero
// (2D tensor). Tracked by TENSORcnt.
__device__ __forceinline__ void tdm_load_2d(
    const void* gptr, unsigned lds_byte, unsigned tensor_d0, unsigned tensor_d1,
    unsigned tile_d0, unsigned tile_d1, unsigned stride0_elems) {
  unsigned long long ga = (unsigned long long)(uintptr_t)gptr;
  u32x4 g0;
  g0.x = 1u;                                                  // count=1 (user)
  g0.y = lds_byte;                                            // lds_addr
  g0.z = (unsigned)(ga & 0xffffffffull);                      // global_addr lo
  g0.w = (unsigned)((ga >> 32) & 0x01ffffffu) | (2u << 30);   // hi | type=2
  i32x8 g1;
  g1[0] = (int)(1u << 16);                                    // data_size=2B
  g1[1] = (int)((tensor_d0 & 0xffffu) << 16);                 // dim0[15:0]
  g1[2] = (int)(((tensor_d0 >> 16) & 0xffffu) |
                ((tensor_d1 & 0xffffu) << 16));               // dim0 hi|dim1 lo
  g1[3] = (int)(((tensor_d1 >> 16) & 0xffffu) |
                ((tile_d0 & 0xffffu) << 16));                 // dim1 hi|tile0
  g1[4] = (int)(tile_d1 & 0xffffu);                           // tile1, tile2=0
  g1[5] = (int)stride0_elems;                                 // dim0 stride lo
  g1[6] = 0;
  g1[7] = 0;
  i32x4 z4 = {};
#if TDM_SIX_ARGS
  i32x8 z8 = {};
  __builtin_amdgcn_tensor_load_to_lds(g0, g1, z4, z4, z8, 0);
#else
  __builtin_amdgcn_tensor_load_to_lds(g0, g1, z4, z4, 0);
#endif
}
#endif  // USE_TDM

// ---------------------------------------------------------------------------
// f32 -> bf16 elementwise conversion (weights)
// ---------------------------------------------------------------------------
__global__ __launch_bounds__(256) void cvt_f32_to_bf16(
    const float* __restrict__ in, bf16_t* __restrict__ out, int n) {
  int i = (blockIdx.x * 256 + threadIdx.x) * 4;
  if (i + 3 < n) {
    float4 f = *(const float4*)(in + i);
    out[i + 0] = f2bf(f.x);
    out[i + 1] = f2bf(f.y);
    out[i + 2] = f2bf(f.z);
    out[i + 3] = f2bf(f.w);
  }
}

// ---------------------------------------------------------------------------
// LayerNorm over C=1024 columns, one row per block, bf16 output
// ---------------------------------------------------------------------------
__global__ __launch_bounds__(256) void layernorm_to_bf16(
    const float* __restrict__ x, const float* __restrict__ g,
    const float* __restrict__ be, bf16_t* __restrict__ out, int C) {
  __shared__ float s1[256];
  __shared__ float s2[256];
  const int row = blockIdx.x, tid = threadIdx.x;
  const float* xr = x + (size_t)row * C;
  float a[4];
  float sum = 0.f, sq = 0.f;
#pragma unroll
  for (int i = 0; i < 4; i++) {
    a[i] = xr[tid + i * 256];
    sum += a[i];
    sq += a[i] * a[i];
  }
  s1[tid] = sum; s2[tid] = sq;
  __syncthreads();
  for (int s = 128; s > 0; s >>= 1) {
    if (tid < s) { s1[tid] += s1[tid + s]; s2[tid] += s2[tid + s]; }
    __syncthreads();
  }
  const float mu = s1[0] * (1.0f / 1024.0f);
  const float var = s2[0] * (1.0f / 1024.0f) - mu * mu;
  const float rstd = rsqrtf(var + 1e-5f);
  bf16_t* orow = out + (size_t)row * C;
#pragma unroll
  for (int i = 0; i < 4; i++) {
    const int c = tid + i * 256;
    orow[c] = f2bf((a[i] - mu) * rstd * g[c] + be[c]);
  }
}

// ---------------------------------------------------------------------------
// bf16 GEMM, f32 accumulation via v_wmma_f32_16x16x32_bf16.
// Block tile 128x128x32, 256 threads = 8 waves, each wave owns 32x64 (2x4
// WMMA tiles). A tile staged in LDS by the Tensor Data Mover when available
// (wave 0 issues, s_wait_tensorcnt 0, barrier publishes), else cooperatively.
// B staged transposed [n][k] so B-fragments are two contiguous b128 reads.
// EPI: 0 = f32 out + residual, 1 = bf16 out, 2 = exact GELU -> bf16 out.
// ---------------------------------------------------------------------------
#define GBM 128
#define GBN 128
#define GBK 32
#if USE_TDM
#define AS_LD 32   // TDM writes the tile compact
#else
#define AS_LD 40   // padded to rotate LDS banks
#endif
#define BS_LD 40

template <int EPI>
__global__ __launch_bounds__(256) void gemm_bf16_wmma(
    const bf16_t* __restrict__ A, const bf16_t* __restrict__ Bm,
    const float* __restrict__ bias, const float* __restrict__ resid,
    float* __restrict__ outF, bf16_t* __restrict__ outB,
    int M, int N, int K) {
  __shared__ __align__(16) bf16_t As[GBM * AS_LD];
  __shared__ __align__(16) bf16_t Bs[GBN * BS_LD];

  const int tid = threadIdx.x;
  const int wave = tid >> 5;
  const int lane = tid & 31;
  const int hf = lane >> 4;
  const int l16 = lane & 15;

  const int bm = blockIdx.y * GBM;
  const int bn = blockIdx.x * GBN;

  const int ar = tid >> 1;             // A row within tile (0..127)
  const int ac = (tid & 1) * 16;       // A col base (0 or 16)
  const int bkr = tid >> 3;            // B k row (0..31)
  const int nbc = (tid & 7) * 16;      // B col base (0..112)

  const int wm = (wave & 3) * 32;
  const int wn = (wave >> 2) * 64;

#if USE_TDM
  // LDS aperture flat address truncates to the LDS byte offset (ISA 10.2)
  const unsigned ldsA = (unsigned)(uintptr_t)(void*)&As[0];
#endif

  v8f acc[2][4] = {};

  for (int k0 = 0; k0 < K; k0 += GBK) {
#if USE_TDM
    if (wave == 0) {
      // A tile: 128 rows x 32 cols, row stride K elements
      tdm_load_2d(A + (size_t)bm * K + k0, ldsA,
                  (unsigned)K, (unsigned)M, GBK, GBM, (unsigned)K);
    }
#else
    {
      const uint4* g = (const uint4*)(A + (size_t)(bm + ar) * K + k0 + ac);
      uint4 v0 = g[0];
      uint4 v1 = g[1];
      uint4* s = (uint4*)(&As[ar * AS_LD + ac]);
      s[0] = v0;
      s[1] = v1;
    }
#endif
    // B tile: 32x128, stored transposed [n][k] for contiguous B-fragments
    {
      PackE u;
      u.p = *(const Pack32*)(Bm + (size_t)(k0 + bkr) * N + bn + nbc);
#pragma unroll
      for (int i = 0; i < 16; i++) Bs[(nbc + i) * BS_LD + bkr] = u.e[i];
    }
    // prefetch next K-step tiles into GL2 while this step computes
    if (k0 + GBK < K) {
      __builtin_prefetch(A + (size_t)(bm + ar) * K + k0 + GBK + ac, 0, 0);
      __builtin_prefetch(Bm + (size_t)(k0 + GBK + bkr) * N + bn + nbc, 0, 0);
    }
#if USE_TDM
    if (wave == 0) __builtin_amdgcn_s_wait_tensorcnt(0);
#endif
    __syncthreads();

    v16bf af[2], bfb[4];
#pragma unroll
    for (int mt = 0; mt < 2; mt++) {
      const bf16_t* base = &As[(wm + mt * 16 + l16) * AS_LD + hf * 8];
      af[mt] = ld_frag_split(base, base + 16);
    }
#pragma unroll
    for (int nt = 0; nt < 4; nt++) {
      const bf16_t* base = &Bs[(wn + nt * 16 + l16) * BS_LD + hf * 8];
      bfb[nt] = ld_frag_split(base, base + 16);
    }
#pragma unroll
    for (int mt = 0; mt < 2; mt++)
#pragma unroll
      for (int nt = 0; nt < 4; nt++)
        acc[mt][nt] = __builtin_amdgcn_wmma_f32_16x16x32_bf16(
            false, af[mt], false, bfb[nt], (short)0, acc[mt][nt], false, false);
    __syncthreads();
  }

  // epilogue (compile-time specialized): C layout row = vgpr + 8*half,
  // col = lane&15
#pragma unroll
  for (int mt = 0; mt < 2; mt++) {
#pragma unroll
    for (int nt = 0; nt < 4; nt++) {
      const int col = bn + wn + nt * 16 + l16;
      const float bv = bias[col];
#pragma unroll
      for (int i = 0; i < 8; i++) {
        const int row = bm + wm + mt * 16 + hf * 8 + i;
        const size_t idx = (size_t)row * N + col;
        float v = acc[mt][nt][i] + bv;
        if (EPI == 2) {
          v = 0.5f * v * (1.0f + erff(v * 0.70710678118654752f));
          outB[idx] = f2bf(v);
        } else if (EPI == 1) {
          outB[idx] = f2bf(v);
        } else {
          outF[idx] = v + resid[idx];
        }
      }
    }
  }
}

// ---------------------------------------------------------------------------
// Causal flash attention, head_dim=64, bf16 in/out, f32 softmax state.
// Block = 8 waves, each wave owns a 16-row query tile (128 q rows / block).
// K chunk (64 keys x 64 dims) in LDS as-is = B^T for S = Q K^T.
// V chunk stored transposed [dim][key] = B^T for O += P V.
// P bounced through per-wave LDS scratch to reshape C-layout -> A-fragment.
// ---------------------------------------------------------------------------
#define HD 64
#define KC 64
#define PH_LD 72

__global__ __launch_bounds__(256) void attention_wmma(
    const bf16_t* __restrict__ q, const bf16_t* __restrict__ kk,
    const bf16_t* __restrict__ vv, bf16_t* __restrict__ o, int T, int H) {
  __shared__ __align__(16) bf16_t Ks[KC * PH_LD];
  __shared__ __align__(16) bf16_t Vt[HD * PH_LD];
  __shared__ __align__(16) bf16_t Ps[8 * 16 * PH_LD];

  const int tid = threadIdx.x;
  const int wave = tid >> 5;
  const int lane = tid & 31;
  const int hf = lane >> 4;
  const int l16 = lane & 15;
  const int C = H * HD;  // 1024

  const int bh = blockIdx.y;
  const int b = bh / H, h = bh % H;
  const size_t rowbase = (size_t)b * T;
  const int qb0 = blockIdx.x * 128;
  const int qo = qb0 + wave * 16;

  v16bf qf[2];
  {
    const bf16_t* qrow = q + (rowbase + qo + l16) * C + h * HD + hf * 8;
#pragma unroll
    for (int c = 0; c < 2; c++)
      qf[c] = ld_frag_split(qrow + c * 32, qrow + c * 32 + 16);
  }

  float mrow[8], lrow[8];
  v8f oacc[4] = {};
#pragma unroll
  for (int i = 0; i < 8; i++) { mrow[i] = -3.0e38f; lrow[i] = 0.0f; }

  bf16_t* pw = &Ps[wave * 16 * PH_LD];

  const int ldk = tid >> 2;
  const int ldd = (tid & 3) * 16;

  for (int kb = 0; kb < qb0 + 128; kb += KC) {
    {
      Pack32 pk = *(const Pack32*)(kk + (rowbase + kb + ldk) * C + h * HD + ldd);
      *(Pack32*)(&Ks[ldk * PH_LD + ldd]) = pk;
      PackE uv;
      uv.p = *(const Pack32*)(vv + (rowbase + kb + ldk) * C + h * HD + ldd);
#pragma unroll
      for (int i = 0; i < 16; i++) Vt[(ldd + i) * PH_LD + ldk] = uv.e[i];
    }
    __syncthreads();

    if (kb <= qo + 15) {
      v8f st[4];
#pragma unroll
      for (int kt = 0; kt < 4; kt++) {
        v8f s = {};
#pragma unroll
        for (int c = 0; c < 2; c++) {
          const bf16_t* base = &Ks[(kt * 16 + l16) * PH_LD + c * 32 + hf * 8];
          v16bf bk_ = ld_frag_split(base, base + 16);
          s = __builtin_amdgcn_wmma_f32_16x16x32_bf16(
              false, qf[c], false, bk_, (short)0, s, false, false);
        }
        const int colg = kb + kt * 16 + l16;
#pragma unroll
        for (int i = 0; i < 8; i++) {
          const int rowg = qo + hf * 8 + i;
          st[kt][i] = (colg <= rowg) ? s[i] * 0.125f : -3.0e38f;
        }
      }
      float mn[8], psc[8], rs[8];
#pragma unroll
      for (int i = 0; i < 8; i++) {
        float mt_ = fmaxf(fmaxf(st[0][i], st[1][i]), fmaxf(st[2][i], st[3][i]));
        mt_ = redmax16(mt_);
        mn[i] = fmaxf(mrow[i], mt_);
        psc[i] = __expf(mrow[i] - mn[i]);
        rs[i] = 0.0f;
      }
#pragma unroll
      for (int kt = 0; kt < 4; kt++)
#pragma unroll
        for (int i = 0; i < 8; i++) {
          const float p = __expf(st[kt][i] - mn[i]);
          st[kt][i] = p;
          rs[i] += p;
        }
#pragma unroll
      for (int i = 0; i < 8; i++) {
        rs[i] = redsum16(rs[i]);
        lrow[i] = lrow[i] * psc[i] + rs[i];
        mrow[i] = mn[i];
      }
#pragma unroll
      for (int nt = 0; nt < 4; nt++)
#pragma unroll
        for (int i = 0; i < 8; i++) oacc[nt][i] *= psc[i];

#pragma unroll
      for (int kt = 0; kt < 4; kt++)
#pragma unroll
        for (int i = 0; i < 8; i++)
          pw[(hf * 8 + i) * PH_LD + kt * 16 + l16] = f2bf(st[kt][i]);
      asm volatile("s_wait_dscnt 0" ::: "memory");  // LDS RAW, same wave

#pragma unroll
      for (int c = 0; c < 2; c++) {
        const bf16_t* pbase = &pw[l16 * PH_LD + c * 32 + hf * 8];
        v16bf ap = ld_frag_split(pbase, pbase + 16);
#pragma unroll
        for (int nt = 0; nt < 4; nt++) {
          const bf16_t* vbase = &Vt[(nt * 16 + l16) * PH_LD + c * 32 + hf * 8];
          v16bf bv_ = ld_frag_split(vbase, vbase + 16);
          oacc[nt] = __builtin_amdgcn_wmma_f32_16x16x32_bf16(
              false, ap, false, bv_, (short)0, oacc[nt], false, false);
        }
      }
    }
    __syncthreads();
  }

  float invl[8];
#pragma unroll
  for (int i = 0; i < 8; i++) invl[i] = 1.0f / lrow[i];
#pragma unroll
  for (int nt = 0; nt < 4; nt++)
#pragma unroll
    for (int i = 0; i < 8; i++) {
      const size_t row = rowbase + qo + hf * 8 + i;
      o[row * C + h * HD + nt * 16 + l16] = f2bf(oacc[nt][i] * invl[i]);
    }
}

// ---------------------------------------------------------------------------
// Host-side orchestration
// ---------------------------------------------------------------------------
extern "C" void kernel_launch(void* const* d_in, const int* in_sizes, int n_in,
                              void* d_out, int out_size, void* d_ws, size_t ws_size,
                              hipStream_t stream) {
  (void)in_sizes; (void)n_in; (void)out_size; (void)ws_size;
  constexpr int D = 1024, Bsz = 4, T = 2048, H = 16;
  constexpr int M = Bsz * T;   // 8192
  constexpr int F = 4 * D;     // 4096

  const float* x   = (const float*)d_in[0];
  const float* Wq  = (const float*)d_in[1];
  const float* bq  = (const float*)d_in[2];
  const float* Wk  = (const float*)d_in[3];
  const float* bk  = (const float*)d_in[4];
  const float* Wv  = (const float*)d_in[5];
  const float* bv  = (const float*)d_in[6];
  const float* Wo  = (const float*)d_in[7];
  const float* bo  = (const float*)d_in[8];
  const float* W1  = (const float*)d_in[9];
  const float* b1  = (const float*)d_in[10];
  const float* W2  = (const float*)d_in[11];
  const float* b2  = (const float*)d_in[12];
  const float* g1  = (const float*)d_in[13];
  const float* be1 = (const float*)d_in[14];
  const float* g2  = (const float*)d_in[15];
  const float* be2 = (const float*)d_in[16];
  float* out = (float*)d_out;

  char* ws = (char*)d_ws;
  size_t off = 0;
  auto take = [&](size_t bytes) -> char* {
    char* p = ws + off;
    off += (bytes + 255) & ~(size_t)255;
    return p;
  };
  bf16_t* Wq_b = (bf16_t*)take((size_t)D * D * 2);
  bf16_t* Wk_b = (bf16_t*)take((size_t)D * D * 2);
  bf16_t* Wv_b = (bf16_t*)take((size_t)D * D * 2);
  bf16_t* Wo_b = (bf16_t*)take((size_t)D * D * 2);
  bf16_t* W1_b = (bf16_t*)take((size_t)D * F * 2);
  bf16_t* W2_b = (bf16_t*)take((size_t)F * D * 2);
  bf16_t* xln  = (bf16_t*)take((size_t)M * D * 2);
  bf16_t* qbuf = (bf16_t*)take((size_t)M * D * 2);
  bf16_t* kbuf = (bf16_t*)take((size_t)M * D * 2);
  bf16_t* vbuf = (bf16_t*)take((size_t)M * D * 2);
  bf16_t* attn = (bf16_t*)take((size_t)M * D * 2);
  float*  x1   = (float*)take((size_t)M * D * 4);
  bf16_t* ln2b = (bf16_t*)take((size_t)M * D * 2);
  bf16_t* h1   = (bf16_t*)take((size_t)M * F * 2);

  const dim3 blk(256);

  cvt_f32_to_bf16<<<dim3(D * D / 1024), blk, 0, stream>>>(Wq, Wq_b, D * D);
  cvt_f32_to_bf16<<<dim3(D * D / 1024), blk, 0, stream>>>(Wk, Wk_b, D * D);
  cvt_f32_to_bf16<<<dim3(D * D / 1024), blk, 0, stream>>>(Wv, Wv_b, D * D);
  cvt_f32_to_bf16<<<dim3(D * D / 1024), blk, 0, stream>>>(Wo, Wo_b, D * D);
  cvt_f32_to_bf16<<<dim3(D * F / 1024), blk, 0, stream>>>(W1, W1_b, D * F);
  cvt_f32_to_bf16<<<dim3(F * D / 1024), blk, 0, stream>>>(W2, W2_b, F * D);

  layernorm_to_bf16<<<dim3(M), blk, 0, stream>>>(x, g1, be1, xln, D);

  gemm_bf16_wmma<1><<<dim3(D / GBN, M / GBM), blk, 0, stream>>>(
      xln, Wq_b, bq, nullptr, nullptr, qbuf, M, D, D);
  gemm_bf16_wmma<1><<<dim3(D / GBN, M / GBM), blk, 0, stream>>>(
      xln, Wk_b, bk, nullptr, nullptr, kbuf, M, D, D);
  gemm_bf16_wmma<1><<<dim3(D / GBN, M / GBM), blk, 0, stream>>>(
      xln, Wv_b, bv, nullptr, nullptr, vbuf, M, D, D);

  attention_wmma<<<dim3(T / 128, Bsz * H), blk, 0, stream>>>(
      qbuf, kbuf, vbuf, attn, T, H);

  gemm_bf16_wmma<0><<<dim3(D / GBN, M / GBM), blk, 0, stream>>>(
      attn, Wo_b, bo, x, x1, nullptr, M, D, D);

  layernorm_to_bf16<<<dim3(M), blk, 0, stream>>>(x1, g2, be2, ln2b, D);

  gemm_bf16_wmma<2><<<dim3(F / GBN, M / GBM), blk, 0, stream>>>(
      ln2b, W1_b, b1, nullptr, nullptr, h1, M, F, D);
  gemm_bf16_wmma<0><<<dim3(D / GBN, M / GBM), blk, 0, stream>>>(
      h1, W2_b, b2, x1, out, nullptr, M, D, F);
}
